// Attention_79645873537043
// MI455X (gfx1250) — matrix operand order repaired and checked by reading, not compiled, not run on hardware
//
#include <hip/hip_runtime.h>
#include <hip/hip_bf16.h>

// Problem constants (match reference)
#define BB 4
#define NN 16384
#define MM 16384
#define KNB 16
#define CC 64
#define OO 64
#define GG 8

typedef __attribute__((ext_vector_type(16))) _Float16 v16h;
typedef __attribute__((ext_vector_type(8)))  _Float16 v8h;
typedef __attribute__((ext_vector_type(8)))  float    v8f;

__device__ __forceinline__ v8f wmma_step(v16h a, v16h b, v8f c) {
    // D = A(16x32 f16) x B(32x16 f16) + C(16x16 f32)
    return __builtin_amdgcn_wmma_f32_16x16x32_f16(false, a, false, b, (short)0, c, false, false);
}

// A fragment (16x32 f16) from row-major LDS tile [16 rows][rowStride halves], K window at kOff.
// Wave32 layout: lane L: row = L&15, half = L>>4; elems 0..7 -> K=8*half+j ; elems 8..15 -> K=16+8*half+j
__device__ __forceinline__ v16h load_a(const _Float16* tile, int rowStride, int kOff) {
    int lane = threadIdx.x & 31;
    int row  = lane & 15;
    int half = lane >> 4;
    const _Float16* base = tile + row * rowStride + kOff;
    union { v16h v; v8h h[2]; } u;
    u.h[0] = *(const v8h*)(base + 8 * half);
    u.h[1] = *(const v8h*)(base + 16 + 8 * half);
    return u.v;
}

// B fragment (32x16 f16) from LDS holding B-transposed row-major: wT[ n(col) ][ K ].
// lane L: col = L&15, half = L>>4; elems 0..7 -> K=8*half+j ; elems 8..15 -> K=16+8*half+j
__device__ __forceinline__ v16h load_b(const _Float16* wT, int rowStride, int kOff) {
    int lane = threadIdx.x & 31;
    int col  = lane & 15;
    int half = lane >> 4;
    const _Float16* base = wT + col * rowStride + kOff;
    union { v16h v; v8h h[2]; } u;
    u.h[0] = *(const v8h*)(base + 8 * half);
    u.h[1] = *(const v8h*)(base + 16 + 8 * half);
    return u.v;
}

// ---------------------------------------------------------------------------
// Kernel A1: x_k = fea^T @ Wk + bk ; x_v = fea^T @ Wv + bv.  64 rows/block.
// grid = BB * (NN/64), block = 128 (4 waves); each wave owns a 16-row subtile.
// ---------------------------------------------------------------------------
__global__ __launch_bounds__(128) void kv_proj(const float* __restrict__ fea,
                                               const float* __restrict__ Wk, const float* __restrict__ bk,
                                               const float* __restrict__ Wv, const float* __restrict__ bv,
                                               float* __restrict__ xk, float* __restrict__ xv) {
    const int b  = blockIdx.x >> 8;           // NN/64 = 256
    const int n0 = (blockIdx.x & 255) * 64;
    __shared__ _Float16 wkT[64 * 64];         // Wk transposed: [o][c]
    __shared__ _Float16 wvT[64 * 64];
    __shared__ _Float16 afea[64 * 64];        // A tile: [n_local][c]
    const int t = threadIdx.x;

    for (int i = t; i < 4096; i += 128) {
        int c = i >> 6, o = i & 63;
        wkT[o * 64 + c] = (_Float16)Wk[c * 64 + o];
        wvT[o * 64 + c] = (_Float16)Wv[c * 64 + o];
    }
    const float* feaB = fea + (size_t)b * CC * NN;
    for (int i = t; i < 4096; i += 128) {
        int c = i >> 6, nl = i & 63;          // coalesced over nl
        afea[nl * 64 + c] = (_Float16)feaB[(size_t)c * NN + n0 + nl];
    }
    __syncthreads();

    const int wave = t >> 5, lane = t & 31;
    const int cn = lane & 15, half = lane >> 4;
    const _Float16* arow = &afea[(wave * 16) * 64];
    v16h a0 = load_a(arow, 64, 0);
    v16h a1 = load_a(arow, 64, 32);

    for (int tc = 0; tc < 4; ++tc) {
        const int col = tc * 16 + cn;
        v8f ck, cv;
        float bkc = bk[col], bvc = bv[col];
#pragma unroll
        for (int j = 0; j < 8; ++j) { ck[j] = bkc; cv[j] = bvc; }
        v16h b0 = load_b(&wkT[(tc * 16) * 64], 64, 0);
        v16h b1 = load_b(&wkT[(tc * 16) * 64], 64, 32);
        ck = wmma_step(a0, b0, ck);
        ck = wmma_step(a1, b1, ck);
        v16h d0 = load_b(&wvT[(tc * 16) * 64], 64, 0);
        v16h d1 = load_b(&wvT[(tc * 16) * 64], 64, 32);
        cv = wmma_step(a0, d0, cv);
        cv = wmma_step(a1, d1, cv);
        size_t rowBase = ((size_t)b * NN + n0 + wave * 16 + 8 * half) * 64;
#pragma unroll
        for (int j = 0; j < 8; ++j) {
            xk[rowBase + (size_t)j * 64 + col] = ck[j];
            xv[rowBase + (size_t)j * 64 + col] = cv[j];
        }
    }
}

// ---------------------------------------------------------------------------
// Kernel A2: qc = cpr - x_q + (cpe_b2 - bq)
//   = (-center_fea^T) @ Wq + relu((cpos@cpe_w1)*s+b) @ cpe_w2 + (cpe_b2-bq)
// A tile for Wq is staged NEGATED (f16 WMMA NEG bits are C-only per ISA).
// ---------------------------------------------------------------------------
__global__ __launch_bounds__(128) void qc_proj(const float* __restrict__ center_fea,
                                               const float* __restrict__ center_pos,
                                               const float* __restrict__ Wq, const float* __restrict__ bq,
                                               const float* __restrict__ cpe_w1, const float* __restrict__ cpe_s,
                                               const float* __restrict__ cpe_b,
                                               const float* __restrict__ cpe_w2, const float* __restrict__ cpe_b2,
                                               float* __restrict__ qcw) {
    const int b  = blockIdx.x >> 8;           // MM/64 = 256
    const int m0 = (blockIdx.x & 255) * 64;
    __shared__ _Float16 wqT[64 * 64];
    __shared__ _Float16 w2T[64 * 64];
    __shared__ _Float16 afea[64 * 64];        // negated center_fea tile
    __shared__ _Float16 ah[64 * 64];          // cpr hidden tile
    const int t = threadIdx.x;

    for (int i = t; i < 4096; i += 128) {
        int c = i >> 6, o = i & 63;
        wqT[o * 64 + c] = (_Float16)Wq[c * 64 + o];
        w2T[o * 64 + c] = (_Float16)cpe_w2[c * 64 + o];
    }
    const float* cfB = center_fea + (size_t)b * CC * MM;
    for (int i = t; i < 4096; i += 128) {
        int c = i >> 6, ml = i & 63;
        afea[ml * 64 + c] = (_Float16)(-cfB[(size_t)c * MM + m0 + ml]);
    }
    {   // layer-1 of cpe in VALU (K=3): thread t does half a row (32 outputs)
        int r = t >> 1, off = (t & 1) * 32;
        int m = m0 + r;
        float px = center_pos[((size_t)b * MM + m) * 3 + 0];
        float py = center_pos[((size_t)b * MM + m) * 3 + 1];
        float pz = center_pos[((size_t)b * MM + m) * 3 + 2];
#pragma unroll
        for (int o2 = 0; o2 < 32; ++o2) {
            int o = off + o2;
            float h = (px * cpe_w1[o] + py * cpe_w1[64 + o] + pz * cpe_w1[128 + o]) * cpe_s[o] + cpe_b[o];
            ah[r * 64 + o] = (_Float16)fmaxf(h, 0.f);
        }
    }
    __syncthreads();

    const int wave = t >> 5, lane = t & 31;
    const int cn = lane & 15, half = lane >> 4;
    const _Float16* arow = &afea[(wave * 16) * 64];
    v16h a0 = load_a(arow, 64, 0);
    v16h a1 = load_a(arow, 64, 32);
    const _Float16* hrow = &ah[(wave * 16) * 64];
    v16h h0 = load_a(hrow, 64, 0);
    v16h h1 = load_a(hrow, 64, 32);

    for (int tc = 0; tc < 4; ++tc) {
        const int col = tc * 16 + cn;
        v8f c;
        float bini = cpe_b2[col] - bq[col];
#pragma unroll
        for (int j = 0; j < 8; ++j) c[j] = bini;
        v16h b0 = load_b(&wqT[(tc * 16) * 64], 64, 0);
        v16h b1 = load_b(&wqT[(tc * 16) * 64], 64, 32);
        c = wmma_step(a0, b0, c);           // -x_q
        c = wmma_step(a1, b1, c);
        v16h d0 = load_b(&w2T[(tc * 16) * 64], 64, 0);
        v16h d1 = load_b(&w2T[(tc * 16) * 64], 64, 32);
        c = wmma_step(h0, d0, c);           // + cpr
        c = wmma_step(h1, d1, c);
        size_t rowBase = ((size_t)b * MM + m0 + wave * 16 + 8 * half) * 64;
#pragma unroll
        for (int j = 0; j < 8; ++j)
            qcw[rowBase + (size_t)j * 64 + col] = c[j];
    }
}

// ---------------------------------------------------------------------------
// Kernel B: fused gather + we-MLP (WMMA) + softmax + pe-MLP (WMMA) + reduce.
// Block = 128 threads (4 waves) handles 8 centers x 16 neighbors = 128 rows.
// grid = BB * (MM/8) = 8192.
// ---------------------------------------------------------------------------
__global__ __launch_bounds__(128) void attn_main(const float* __restrict__ center_pos,
                                                 const float* __restrict__ pos,
                                                 const int*   __restrict__ idx,
                                                 const float* __restrict__ xk,
                                                 const float* __restrict__ xv,
                                                 const float* __restrict__ qc,
                                                 const float* __restrict__ pe_w1, const float* __restrict__ pe_s,
                                                 const float* __restrict__ pe_b,  const float* __restrict__ pe_w2,
                                                 const float* __restrict__ pe_b2,
                                                 const float* __restrict__ we_w1, const float* __restrict__ we_s,
                                                 const float* __restrict__ we_b,  const float* __restrict__ we_w2,
                                                 const float* __restrict__ we_b2,
                                                 float* __restrict__ out) {
    const int b  = blockIdx.x >> 11;          // MM/8 = 2048
    const int m0 = (blockIdx.x & 2047) * 8;

    __shared__ _Float16 sA[128 * 64];         // phase 1-2: r_qk (f16); phase 5-6: pe hidden
    __shared__ _Float16 sV[128 * 64];         // v = nv (+ pr), f16
    __shared__ _Float16 sW1[128 * 32];        // we layer-1 output, K-padded to 32 (zeros)
    __shared__ float    sLog[128 * 8];        // we logits -> softmax weights
    __shared__ _Float16 weT1[16 * 64];        // we_w1^T, G padded to 16
    __shared__ _Float16 weT2[16 * 32];        // we_w2^T, padded
    __shared__ _Float16 peT2[64 * 64];        // pe_w2^T

    const int t = threadIdx.x;
    const int wave = t >> 5, lane = t & 31;
    const int cn = lane & 15, half = lane >> 4;

    // ---- Phase 0: stage weights (padded with zeros), zero K-pad of sW1 ----
    for (int i = t; i < 1024; i += 128) {
        int g = i >> 6, c = i & 63;
        weT1[g * 64 + c] = (g < 8) ? (_Float16)we_w1[c * 8 + g] : (_Float16)0.f;
    }
    for (int i = t; i < 512; i += 128) {
        int g = i >> 5, k = i & 31;
        weT2[g * 32 + k] = (g < 8 && k < 8) ? (_Float16)we_w2[k * 8 + g] : (_Float16)0.f;
    }
    for (int i = t; i < 4096; i += 128) {
        int o = i >> 6, c = i & 63;
        peT2[o * 64 + c] = (_Float16)pe_w2[c * 64 + o];
    }
#pragma unroll
    for (int k2 = 16; k2 < 32; ++k2) sW1[t * 32 + k2] = (_Float16)0.f;

    // ---- Phase 1: per-row gather. Row r = t: m_l=r>>4, k=r&15 ----
    const int r = t;
    const int m_l = r >> 4, kk = r & 15;
    const int m = m0 + m_l;
    const int n = idx[((size_t)b * MM + m) * KNB + kk];
    float px, py, pz;
    {
        const float* pn = pos + ((size_t)b * NN + n) * 3;
        const float* pc = center_pos + ((size_t)b * MM + m) * 3;
        px = pn[0] - pc[0]; py = pn[1] - pc[1]; pz = pn[2] - pc[2];
    }
    {
        const float* xkrow = xk + ((size_t)b * NN + n) * 64;
        const float* xvrow = xv + ((size_t)b * NN + n) * 64;
        const float* qcrow = qc + ((size_t)b * MM + m) * 64;
#pragma unroll
        for (int j = 0; j < 64; j += 4) {
            float4 a = *(const float4*)(xkrow + j);
            float4 q = *(const float4*)(qcrow + j);
            float4 v = *(const float4*)(xvrow + j);
            sA[r * 64 + j + 0] = (_Float16)(a.x + q.x);
            sA[r * 64 + j + 1] = (_Float16)(a.y + q.y);
            sA[r * 64 + j + 2] = (_Float16)(a.z + q.z);
            sA[r * 64 + j + 3] = (_Float16)(a.w + q.w);
            sV[r * 64 + j + 0] = (_Float16)v.x;
            sV[r * 64 + j + 1] = (_Float16)v.y;
            sV[r * 64 + j + 2] = (_Float16)v.z;
            sV[r * 64 + j + 3] = (_Float16)v.w;
        }
    }
    __syncthreads();

    // ---- Phase 2: we layer-1: [128,64] @ [64,8->16], scale+bias+relu ----
    for (int q2 = 0; q2 < 2; ++q2) {
        const int rt = wave + q2 * 4;
        v16h a0 = load_a(&sA[rt * 16 * 64], 64, 0);
        v16h a1 = load_a(&sA[rt * 16 * 64], 64, 32);
        v16h b0 = load_b(weT1, 64, 0);
        v16h b1 = load_b(weT1, 64, 32);
        v8f c;
#pragma unroll
        for (int j = 0; j < 8; ++j) c[j] = 0.f;
        c = wmma_step(a0, b0, c);
        c = wmma_step(a1, b1, c);
        float sc = (cn < 8) ? we_s[cn] : 0.f;
        float bc = (cn < 8) ? we_b[cn] : 0.f;
#pragma unroll
        for (int j = 0; j < 8; ++j) {
            int row = rt * 16 + j + 8 * half;
            float v = fmaxf(c[j] * sc + bc, 0.f);
            sW1[row * 32 + cn] = (_Float16)((cn < 8) ? v : 0.f);
        }
    }
    __syncthreads();

    // ---- Phase 3: we layer-2: [128,16(pad32)] @ [16->32,8->16] ----
    for (int q2 = 0; q2 < 2; ++q2) {
        const int rt = wave + q2 * 4;
        v16h a = load_a(&sW1[rt * 16 * 32], 32, 0);
        v16h bw = load_b(weT2, 32, 0);
        v8f c;
        float bini = (cn < 8) ? we_b2[cn] : 0.f;
#pragma unroll
        for (int j = 0; j < 8; ++j) c[j] = bini;
        c = wmma_step(a, bw, c);
#pragma unroll
        for (int j = 0; j < 8; ++j) {
            int row = rt * 16 + j + 8 * half;
            if (cn < 8) sLog[row * 8 + cn] = c[j];
        }
    }
    __syncthreads();

    // ---- Phase 5a: pe layer-1 in VALU (K=3), writes sA (now free) ----
#pragma unroll 4
    for (int o = 0; o < 64; ++o) {
        float h = (px * pe_w1[o] + py * pe_w1[64 + o] + pz * pe_w1[128 + o]) * pe_s[o] + pe_b[o];
        sA[r * 64 + o] = (_Float16)fmaxf(h, 0.f);
    }
    // ---- Phase 4: softmax over neighbors (threads 0..63: 8 m x 8 g) ----
    if (t < 64) {
        const int ml = t >> 3, g = t & 7;
        float mx = -1e30f;
#pragma unroll
        for (int k = 0; k < 16; ++k) mx = fmaxf(mx, sLog[(ml * 16 + k) * 8 + g]);
        float s = 0.f;
#pragma unroll
        for (int k = 0; k < 16; ++k) {
            float e = __expf(sLog[(ml * 16 + k) * 8 + g] - mx);
            sLog[(ml * 16 + k) * 8 + g] = e;
            s += e;
        }
        float inv = 1.f / s;
#pragma unroll
        for (int k = 0; k < 16; ++k) sLog[(ml * 16 + k) * 8 + g] *= inv;
    }
    __syncthreads();

    // ---- Phase 6: pe layer-2: [128,64] @ [64,64], accumulate into sV ----
    for (int jj = 0; jj < 8; ++jj) {
        const int job = wave * 8 + jj;
        const int rt = job >> 2, tc = job & 3;
        v16h a0 = load_a(&sA[rt * 16 * 64], 64, 0);
        v16h a1 = load_a(&sA[rt * 16 * 64], 64, 32);
        v16h b0 = load_b(&peT2[(tc * 16) * 64], 64, 0);
        v16h b1 = load_b(&peT2[(tc * 16) * 64], 64, 32);
        const int o = tc * 16 + cn;
        v8f c;
        float bini = pe_b2[o];
#pragma unroll
        for (int j = 0; j < 8; ++j) c[j] = bini;
        c = wmma_step(a0, b0, c);
        c = wmma_step(a1, b1, c);
#pragma unroll
        for (int j = 0; j < 8; ++j) {
            int row = rt * 16 + j + 8 * half;
            int a_ = row * 64 + o;
            sV[a_] = (_Float16)((float)sV[a_] + c[j]);
        }
    }
    __syncthreads();

    // ---- Phase 7: weighted reduction over K=16; out is [B, O, M] ----
#pragma unroll
    for (int i = 0; i < 4; ++i) {
        int task = t + i * 128;               // 8 m x 64 o = 512 tasks
        int ml = task >> 6, o = task & 63, g = o >> 3;
        float acc = 0.f;
#pragma unroll
        for (int k = 0; k < 16; ++k)
            acc += (float)sV[(ml * 16 + k) * 64 + o] * sLog[(ml * 16 + k) * 8 + g];
        out[((size_t)b * OO + o) * MM + m0 + ml] = acc;
    }
}

// ---------------------------------------------------------------------------
extern "C" void kernel_launch(void* const* d_in, const int* in_sizes, int n_in,
                              void* d_out, int out_size, void* d_ws, size_t ws_size,
                              hipStream_t stream) {
    const float* center_pos = (const float*)d_in[0];
    const float* center_fea = (const float*)d_in[1];
    const float* pos        = (const float*)d_in[2];
    const float* fea        = (const float*)d_in[3];
    const int*   idx        = (const int*)d_in[4];
    const float* Wq = (const float*)d_in[5];  const float* bq = (const float*)d_in[6];
    const float* Wk = (const float*)d_in[7];  const float* bk = (const float*)d_in[8];
    const float* Wv = (const float*)d_in[9];  const float* bv = (const float*)d_in[10];
    const float* cpe_w1 = (const float*)d_in[11]; const float* cpe_s = (const float*)d_in[12];
    const float* cpe_b  = (const float*)d_in[13]; const float* cpe_w2 = (const float*)d_in[14];
    const float* cpe_b2 = (const float*)d_in[15];
    const float* pe_w1 = (const float*)d_in[16]; const float* pe_s = (const float*)d_in[17];
    const float* pe_b  = (const float*)d_in[18]; const float* pe_w2 = (const float*)d_in[19];
    const float* pe_b2 = (const float*)d_in[20];
    const float* we_w1 = (const float*)d_in[21]; const float* we_s = (const float*)d_in[22];
    const float* we_b  = (const float*)d_in[23]; const float* we_w2 = (const float*)d_in[24];
    const float* we_b2 = (const float*)d_in[25];
    float* out = (float*)d_out;

    // workspace: x_k, x_v (B*N*O) + qc (B*M*O), all f32
    float* xk = (float*)d_ws;
    float* xv = xk + (size_t)BB * NN * OO;
    float* qcw = xv + (size_t)BB * NN * OO;

    kv_proj<<<BB * (NN / 64), 128, 0, stream>>>(fea, Wk, bk, Wv, bv, xk, xv);
    qc_proj<<<BB * (MM / 64), 128, 0, stream>>>(center_fea, center_pos, Wq, bq,
                                                cpe_w1, cpe_s, cpe_b, cpe_w2, cpe_b2, qcw);
    attn_main<<<BB * (MM / 8), 128, 0, stream>>>(center_pos, pos, idx, xk, xv, qcw,
                                                 pe_w1, pe_s, pe_b, pe_w2, pe_b2,
                                                 we_w1, we_s, we_b, we_w2, we_b2, out);
    (void)in_sizes; (void)n_in; (void)out_size; (void)ws_size;
}
